// HybridCNNQuantumModel_34351148434288
// MI455X (gfx1250) — matrix operand order repaired and checked by reading, not compiled, not run on hardware
//
#include <hip/hip_runtime.h>
#include <hip/hip_bf16.h>

typedef __attribute__((ext_vector_type(16))) _Float16 v16h;
typedef __attribute__((ext_vector_type(8)))  _Float16 v8h;
typedef __attribute__((ext_vector_type(8)))  float    v8f;

#define N_POOL1 (13*13*16)   /* 2704 halves per image */
#define KPAD 160             /* 144 real K, padded to 5 chunks of 32 */
#define KC 5

// ---------------------------------------------------------------------------
// Kernel 0: repack conv2 weights [3,3,16,32] f32 -> [kc][n][kk] f16 (K padded)
// B-fragment layout: lane n reads contiguous 16 halves at [kc][n][hi*16].
// ---------------------------------------------------------------------------
__global__ __launch_bounds__(256) void k_prep_w2(const float* __restrict__ w2,
                                                 _Float16* __restrict__ w2h)
{
    for (int i = threadIdx.x; i < KC*32*32; i += 256) {
        int kc  = i >> 10;
        int rem = i & 1023;
        int n   = rem >> 5;
        int kk  = rem & 31;
        int k   = kc*32 + kk;                    // (dy*3+dx)*16 + ci
        float v = (k < 144) ? w2[k*32 + n] : 0.f;
        w2h[i]  = (_Float16)v;
    }
}

// ---------------------------------------------------------------------------
// Kernel 1: conv1 (3x3, 1->16, VALID) + ReLU + 2x2 maxpool -> fp16 [B,13,13,16]
// One block per image; channel is constant per thread (stride 256 ≡ 0 mod 16).
// ---------------------------------------------------------------------------
__global__ __launch_bounds__(256) void k_conv1_pool(const float* __restrict__ x,
                                                    const float* __restrict__ w1,
                                                    const float* __restrict__ b1,
                                                    _Float16* __restrict__ pooled1)
{
    __shared__ float img[784];
    const int tid = threadIdx.x;
    const size_t base = (size_t)blockIdx.x * 784;
    for (int i = tid; i < 784; i += 256) img[i] = x[base + i];
    __syncthreads();

    const int c = tid & 15;
    float wc[9];
#pragma unroll
    for (int k = 0; k < 9; ++k) wc[k] = w1[k*16 + c];
    const float bc = b1[c];

    _Float16* outp = pooled1 + (size_t)blockIdx.x * N_POOL1;
    for (int idx = tid; idx < N_POOL1; idx += 256) {
        int pix = idx >> 4;
        int py = pix / 13, px = pix - 13*py;
        float m = 0.f;                            // post-ReLU values are >= 0
#pragma unroll
        for (int wy = 0; wy < 2; ++wy)
#pragma unroll
            for (int wx = 0; wx < 2; ++wx) {
                int y = 2*py + wy, xx = 2*px + wx;
                float acc = bc;
#pragma unroll
                for (int dy = 0; dy < 3; ++dy)
#pragma unroll
                    for (int dx = 0; dx < 3; ++dx)
                        acc = fmaf(img[(y+dy)*28 + xx+dx], wc[dy*3+dx], acc);
                m = fmaxf(m, acc);                // relu folded into max(0,..)
            }
        outp[idx] = (_Float16)m;
    }
}

// ---------------------------------------------------------------------------
// Kernel 2: conv2 as implicit GEMM via v_wmma_f32_16x16x32_f16
//   A = im2col patches [128(121) x 160(144)] f16 in LDS
//   B = repacked weights [5][32][32] f16 in LDS
// then ReLU + bias, 2x2 maxpool (11->5), dense 800x4 with LDS tree reduction,
// tanh -> feats[image][4].  One block (8 waves) per image.
// ---------------------------------------------------------------------------
__global__ __launch_bounds__(256) void k_conv2_dense(const _Float16* __restrict__ pooled1,
                                                     const _Float16* __restrict__ w2h,
                                                     const float* __restrict__ b2,
                                                     const float* __restrict__ dw,
                                                     const float* __restrict__ db,
                                                     float* __restrict__ feats)
{
    __shared__ alignas(64) _Float16 Aim[128*KPAD];   // 40960 B
    __shared__ alignas(64) _Float16 Bl[KC*32*32];    // 10240 B
    const int tid = threadIdx.x;
    const _Float16* p1 = pooled1 + (size_t)blockIdx.x * N_POOL1;

    for (int i = tid; i < KC*32*32; i += 256) Bl[i] = w2h[i];
    // build im2col tile (zero-padded rows 121..127 and K 144..159)
    for (int i = tid; i < 128*KPAD; i += 256) {
        int p = i / KPAD;
        int k = i - p*KPAD;
        _Float16 v = (_Float16)0.f;
        if (p < 121 && k < 144) {
            int dy  = k / 48;
            int rem = k - 48*dy;
            int dx  = rem >> 4;
            int cc  = rem & 15;
            int oy  = p / 11, ox = p - 11*oy;
            v = p1[((oy+dy)*13 + (ox+dx))*16 + cc];
        }
        Aim[i] = v;
    }
    __syncthreads();

    const int lane = tid & 31;
    const int wv   = tid >> 5;        // wave = M-tile (rows wv*16 .. wv*16+15)
    const int hi   = lane >> 4;
    const int lo16 = lane & 15;
    const int row  = wv*16 + lo16;

    v8f acc0 = {}; v8f acc1 = {};
#pragma unroll
    for (int kc = 0; kc < KC; ++kc) {
        // A fragment: row (lane%16); halves K = kc*32 + hi*16 .. +15
        v8h alo = *(const v8h*)&Aim[row*KPAD + kc*32 + hi*16];
        v8h ahi = *(const v8h*)&Aim[row*KPAD + kc*32 + hi*16 + 8];
        v16h a;
#pragma unroll
        for (int e = 0; e < 8; ++e) { a[e] = alo[e]; a[e+8] = ahi[e]; }
        // B fragments: column n = lane%16 (+16 for second N-tile), K = hi*16..+15
        v16h b0 = *(const v16h*)&Bl[kc*1024 + lo16*32 + hi*16];
        v16h b1 = *(const v16h*)&Bl[kc*1024 + (16 + lo16)*32 + hi*16];
        acc0 = __builtin_amdgcn_wmma_f32_16x16x32_f16(false, a, false, b0,
                                                      (short)0, acc0, false, false);
        acc1 = __builtin_amdgcn_wmma_f32_16x16x32_f16(false, a, false, b1,
                                                      (short)0, acc1, false, false);
    }
    __syncthreads();                  // all waves done reading Aim

    // Reuse Aim region: Cout[121*32] f32 then reduction buffer [256*4] f32.
    float* Cout = (float*)Aim;
    float* red  = Cout + 121*32;
    const float bz0 = b2[lo16], bz1 = b2[16 + lo16];
#pragma unroll
    for (int r = 0; r < 8; ++r) {     // C layout: lanes 16-31 hold M = 8+r
        int p = wv*16 + hi*8 + r;
        if (p < 121) {
            Cout[p*32 + lo16]      = fmaxf(acc0[r] + bz0, 0.f);
            Cout[p*32 + 16 + lo16] = fmaxf(acc1[r] + bz1, 0.f);
        }
    }
    __syncthreads();

    // maxpool 11->5 and dense partials (flat index matches reference reshape)
    float part[4] = {0.f, 0.f, 0.f, 0.f};
    for (int idx = tid; idx < 800; idx += 256) {
        int py  = idx / 160;
        int rem = idx - py*160;
        int px  = rem >> 5;
        int cc  = rem & 31;
        int r0  = (2*py)*11 + 2*px;
        int r1  = r0 + 11;
        float m = fmaxf(fmaxf(Cout[r0*32 + cc], Cout[(r0+1)*32 + cc]),
                        fmaxf(Cout[r1*32 + cc], Cout[(r1+1)*32 + cc]));
#pragma unroll
        for (int j = 0; j < 4; ++j) part[j] = fmaf(m, dw[idx*4 + j], part[j]);
    }
#pragma unroll
    for (int j = 0; j < 4; ++j) red[tid*4 + j] = part[j];
    __syncthreads();
    for (int s = 128; s >= 1; s >>= 1) {   // deterministic tree reduction
        if (tid < s) {
#pragma unroll
            for (int j = 0; j < 4; ++j) red[tid*4 + j] += red[(tid+s)*4 + j];
        }
        __syncthreads();
    }
    if (tid < 4)
        feats[(size_t)blockIdx.x*4 + tid] = tanhf(red[tid] + db[tid]);
}

// ---------------------------------------------------------------------------
// Kernel 3: quantum circuit, analytic closed form.
//   <Z_j> = -sum_b [prod_i (1 + (2b_i-1) sin f_i)/2] * sin(sum_i b_i w_ij)
// then sigmoid(10 * <Z_j>).  One thread per image.
// ---------------------------------------------------------------------------
__global__ __launch_bounds__(256) void k_quantum(const float* __restrict__ feats,
                                                 const float* __restrict__ qw,
                                                 float* __restrict__ out, int B)
{
    int img = blockIdx.x*256 + threadIdx.x;
    if (img >= B) return;
    float p1a[4], p0a[4];
#pragma unroll
    for (int i = 0; i < 4; ++i) {
        float s = __sinf(feats[(size_t)img*4 + i]);
        p1a[i] = 0.5f*(1.f + s);
        p0a[i] = 0.5f*(1.f - s);
    }
#pragma unroll
    for (int j = 0; j < 4; ++j) {
        float q = 0.f;
#pragma unroll
        for (int b = 0; b < 16; ++b) {
            float pb = 1.f, ang = 0.f;
#pragma unroll
            for (int i = 0; i < 4; ++i) {
                if (b & (1 << i)) { pb *= p1a[i]; ang += qw[i*4 + j]; }
                else              { pb *= p0a[i]; }
            }
            q -= pb * __sinf(ang);
        }
        out[(size_t)img*4 + j] = 1.f / (1.f + __expf(-10.f*q));
    }
}

// ---------------------------------------------------------------------------
extern "C" void kernel_launch(void* const* d_in, const int* in_sizes, int n_in,
                              void* d_out, int out_size, void* d_ws, size_t ws_size,
                              hipStream_t stream)
{
    const float* x  = (const float*)d_in[0];
    const float* w1 = (const float*)d_in[1];
    const float* b1 = (const float*)d_in[2];
    const float* w2 = (const float*)d_in[3];
    const float* b2 = (const float*)d_in[4];
    const float* dw = (const float*)d_in[5];
    const float* db = (const float*)d_in[6];
    const float* qw = (const float*)d_in[7];
    float* out = (float*)d_out;
    const int B = in_sizes[0] / (28*28);

    char* ws = (char*)d_ws;
    _Float16* pooled1 = (_Float16*)ws;                       // B*2704 halves
    size_t off = (size_t)B * N_POOL1 * sizeof(_Float16);
    off = (off + 255) & ~(size_t)255;
    _Float16* w2h = (_Float16*)(ws + off);                   // 5120 halves
    off += (size_t)KC*32*32*sizeof(_Float16);
    off = (off + 255) & ~(size_t)255;
    float* feats = (float*)(ws + off);                       // B*4 floats

    k_prep_w2   <<<1,            256, 0, stream>>>(w2, w2h);
    k_conv1_pool<<<B,            256, 0, stream>>>(x, w1, b1, pooled1);
    k_conv2_dense<<<B,           256, 0, stream>>>(pooled1, w2h, b2, dw, db, feats);
    k_quantum   <<<(B + 255)/256,256, 0, stream>>>(feats, qw, out, B);

    (void)n_in; (void)out_size; (void)ws_size;
}
